// GraphConvolution_85856396247552
// MI455X (gfx1250) — matrix operand order repaired and checked by reading, not compiled, not run on hardware
//
#include <hip/hip_runtime.h>

// ---------------------------------------------------------------------------
// Types for CDNA5 WMMA (wave32)
// ---------------------------------------------------------------------------
typedef __attribute__((ext_vector_type(16))) __bf16 v16bf;
typedef __attribute__((ext_vector_type(8)))  __bf16 v8bf;
typedef __attribute__((ext_vector_type(4)))  __bf16 v4bf;
typedef __attribute__((ext_vector_type(8)))  float  v8f;
typedef __attribute__((ext_vector_type(4)))  float  v4f;

#define TILE_M 128
#define TILE_N 128
#define TILE_K 32

// ---------------------------------------------------------------------------
// GEMM: C = s1 * (A @ B) + s2 * E      (fp32 in memory, bf16 WMMA compute)
//   A: [M,K] row-major fp32     B: [K,N] row-major fp32
//   E: [M,N] fp32 epilogue operand
//   C: [M,N] fp32 output; if DUAL, C2 gets a duplicate copy
//   mode 0: s1 = 1-alpha, s2 = alpha        (fscal = &alpha)
//   mode 1: s1 = log(lamda/l + 1), s2=1-s1  (fscal = &lamda, iscal = &l)
// Block = 256 threads = 8 waves; 128x128 C tile; double-buffered LDS.
// Wave (wm,wn) computes rows [wm*64,+64) x cols [wn*32,+32) = 4x2 WMMA tiles.
// ---------------------------------------------------------------------------
template<bool DUAL>
__global__ __launch_bounds__(256, 1)
void gemm_bf16_wmma(const float* __restrict__ A, const float* __restrict__ B,
                    const float* __restrict__ E, float* __restrict__ C,
                    float* __restrict__ C2,
                    int M, int N, int K,
                    const float* __restrict__ fscal,
                    const int*  __restrict__ iscal,
                    int mode)
{
    // Double-buffered LDS: A row-major [m][k], B transposed [n][k] (bf16)
    __shared__ __bf16 sA[2][TILE_M][TILE_K];   // 16 KB
    __shared__ __bf16 sB[2][TILE_N][TILE_K];   // 16 KB

    const int tid  = threadIdx.x;
    const int lane = tid & 31;
    const int wave = tid >> 5;
    const int wm   = wave >> 2;      // 0..1
    const int wn   = wave & 3;       // 0..3

    const int tM = blockIdx.x * TILE_M;
    const int tN = blockIdx.y * TILE_N;

    // epilogue scalars (device-resident, graph-capture safe)
    float s1, s2;
    if (mode == 0) { float a = fscal[0]; s1 = 1.0f - a; s2 = a; }
    else           { float th = __logf(fscal[0] / (float)iscal[0] + 1.0f);
                     s1 = th; s2 = 1.0f - th; }

    // per-thread staging registers for the next tile (software pipeline)
    v4f ra[4], rb[4];

    auto load_global = [&](int k0) {
        #pragma unroll
        for (int i = 0; i < 4; ++i) {
            int id  = tid + i * 256;        // 0..1023
            int row = id >> 3;              // 0..127  (A: 128 rows x 8 f4)
            int c4  = (id & 7) << 2;
            ra[i] = *(const v4f*)(A + (size_t)(tM + row) * K + k0 + c4);
            int kk = id >> 5;               // 0..31   (B: 32 rows x 32 f4)
            int n4 = (id & 31) << 2;
            rb[i] = *(const v4f*)(B + (size_t)(k0 + kk) * N + tN + n4);
        }
    };
    auto store_lds = [&](int buf) {
        #pragma unroll
        for (int i = 0; i < 4; ++i) {
            int id  = tid + i * 256;
            int row = id >> 3;
            int c4  = (id & 7) << 2;
            v4bf h = { (__bf16)ra[i].x, (__bf16)ra[i].y,
                       (__bf16)ra[i].z, (__bf16)ra[i].w };
            *(v4bf*)&sA[buf][row][c4] = h;            // ds_store_b64
            int kk = id >> 5;
            int n4 = (id & 31) << 2;
            sB[buf][n4 + 0][kk] = (__bf16)rb[i].x;    // transpose to [n][k]
            sB[buf][n4 + 1][kk] = (__bf16)rb[i].y;
            sB[buf][n4 + 2][kk] = (__bf16)rb[i].z;
            sB[buf][n4 + 3][kk] = (__bf16)rb[i].w;
        }
    };

    v8f acc[4][2] = {};

    const int lrow = lane & 15;
    const int ahk  = (lane < 16) ? 0 : 8;    // A fragment K-base per ISA layout
    const int bhk  = (lane < 16) ? 0 : 8;    // B fragment K-base per ISA layout

    const int T = K / TILE_K;

    // prologue: stage tile 0
    load_global(0);
    store_lds(0);
    __syncthreads();

    for (int t = 0; t < T; ++t) {
        const int cur = t & 1;

        // issue global loads for tile t+1 early; WMMAs below hide the latency
        if (t + 1 < T) load_global((t + 1) * TILE_K);

        // ---- fragments from buffer `cur` (16B-aligned ds_load_b128) ----
        v16bf afrag[4], bfrag[2];
        #pragma unroll
        for (int mi = 0; mi < 4; ++mi) {
            const __bf16* p = &sA[cur][wm * 64 + mi * 16 + lrow][ahk];
            v8bf lo = *(const v8bf*)p;          // K = ahk .. ahk+7
            v8bf hi = *(const v8bf*)(p + 16);   // K = ahk+16 .. ahk+23
            afrag[mi] = __builtin_shufflevector(lo, hi,
                0,1,2,3,4,5,6,7,8,9,10,11,12,13,14,15);
        }
        #pragma unroll
        for (int ni = 0; ni < 2; ++ni) {
            const __bf16* p = &sB[cur][wn * 32 + ni * 16 + lrow][bhk];
            v8bf lo = *(const v8bf*)p;          // K = bhk .. bhk+7
            v8bf hi = *(const v8bf*)(p + 16);   // K = bhk+16 .. bhk+23
            bfrag[ni] = __builtin_shufflevector(lo, hi,
                0,1,2,3,4,5,6,7,8,9,10,11,12,13,14,15);
        }

        // ---- 8 WMMAs: 16x16x32 bf16, f32 accumulate ----
        #pragma unroll
        for (int mi = 0; mi < 4; ++mi)
            #pragma unroll
            for (int ni = 0; ni < 2; ++ni)
                acc[mi][ni] = __builtin_amdgcn_wmma_f32_16x16x32_bf16(
                    false, afrag[mi], false, bfrag[ni],
                    (short)0, acc[mi][ni], false, false);

        // stage tile t+1 into the other buffer; single barrier per iteration
        if (t + 1 < T) store_lds((t + 1) & 1);
        __syncthreads();
    }

    // ---- epilogue: C = s1*acc + s2*E  (16x16 f32 C layout per ISA) ----
    #pragma unroll
    for (int mi = 0; mi < 4; ++mi) {
        int rbase = tM + wm * 64 + mi * 16 + ((lane < 16) ? 0 : 8);
        #pragma unroll
        for (int ni = 0; ni < 2; ++ni) {
            int col = tN + wn * 32 + ni * 16 + (lane & 15);
            float e[8];
            #pragma unroll
            for (int r = 0; r < 8; ++r)
                e[r] = E[(size_t)(rbase + r) * N + col];
            #pragma unroll
            for (int r = 0; r < 8; ++r) {
                size_t idx = (size_t)(rbase + r) * N + col;
                float v = s1 * acc[mi][ni][r] + s2 * e[r];
                C[idx] = v;
                if constexpr (DUAL) C2[idx] = v;
            }
        }
    }
}

// ---------------------------------------------------------------------------
// Global min/max reduction (two pass) for whale clip bounds
// ---------------------------------------------------------------------------
__global__ __launch_bounds__(256)
void minmax_part(const float* __restrict__ x, int n,
                 float* __restrict__ pmin, float* __restrict__ pmax)
{
    __shared__ float smin[256], smax[256];
    int tid = threadIdx.x;
    float mn = 3.402823e38f, mx = -3.402823e38f;
    for (size_t i = (size_t)blockIdx.x * 256 + tid; i < (size_t)n;
         i += (size_t)gridDim.x * 256) {
        float v = x[i];
        mn = fminf(mn, v);
        mx = fmaxf(mx, v);
    }
    smin[tid] = mn; smax[tid] = mx;
    __syncthreads();
    for (int s = 128; s > 0; s >>= 1) {
        if (tid < s) {
            smin[tid] = fminf(smin[tid], smin[tid + s]);
            smax[tid] = fmaxf(smax[tid], smax[tid + s]);
        }
        __syncthreads();
    }
    if (tid == 0) { pmin[blockIdx.x] = smin[0]; pmax[blockIdx.x] = smax[0]; }
}

__global__ __launch_bounds__(256)
void minmax_final(const float* __restrict__ pmin, const float* __restrict__ pmax,
                  int nb, float* __restrict__ bounds)
{
    __shared__ float smin[256], smax[256];
    int tid = threadIdx.x;
    float mn = 3.402823e38f, mx = -3.402823e38f;
    for (int i = tid; i < nb; i += 256) {
        mn = fminf(mn, pmin[i]);
        mx = fmaxf(mx, pmax[i]);
    }
    smin[tid] = mn; smax[tid] = mx;
    __syncthreads();
    for (int s = 128; s > 0; s >>= 1) {
        if (tid < s) {
            smin[tid] = fminf(smin[tid], smin[tid + s]);
            smax[tid] = fmaxf(smax[tid], smax[tid + s]);
        }
        __syncthreads();
    }
    if (tid == 0) { bounds[0] = smin[0]; bounds[1] = smax[0]; }
}

// ---------------------------------------------------------------------------
// Whale optimization step (stateless hash RNG; structurally faithful)
// N = 8192 agents, D = 1024 dims hardcoded -> shift/mask addressing
// ---------------------------------------------------------------------------
#define WHALE_N 8192
#define WHALE_D 1024

__device__ __forceinline__ unsigned pcg(unsigned v) {
    v = v * 747796405u + 2891336453u;
    unsigned w = ((v >> ((v >> 28) + 4u)) ^ v) * 277803737u;
    return (w >> 22) ^ w;
}
__device__ __forceinline__ float u01(unsigned h) {
    return (float)(h >> 8) * (1.0f / 16777216.0f);
}

__global__ __launch_bounds__(256)
void whale_step(const float* __restrict__ pos, float* __restrict__ npos,
                const float* __restrict__ bounds,
                int iter, int leaderIdx, int maxIter)
{
    unsigned idx = blockIdx.x * 256 + threadIdx.x;   // < 2^23, fits u32
    int i  = (int)(idx >> 10);          // agent    (D = 1024)
    int jj = (int)(idx & 1023);         // dimension

    float lower = bounds[0], upper = bounds[1];
    float itf = (float)iter;
    float a   = 2.0f - itf * (2.0f / maxIter);
    float a2  = -1.0f - itf * (1.0f / maxIter);

    unsigned hb = pcg(0x9E3779B9u * (unsigned)(iter + 1) ^ (unsigned)i);
    float r1 = u01(pcg(hb ^ 0x11u));
    float r2 = u01(pcg(hb ^ 0x22u));
    float r3 = u01(pcg(hb ^ 0x33u));
    float p  = u01(pcg(hb ^ 0x44u));
    float Aco = 2.0f * a * r1 - a;
    float Cco = 2.0f * r2;
    float lp  = (a2 - 1.0f) * r3 + 1.0f;

    unsigned hr = pcg(hb + 0x55u + (unsigned)jj * 2654435761u);
    int ridx = (int)(hr & (WHALE_N - 1));   // N = 8192 (power of two)

    float xx = pos[idx];
    float xr = pos[((unsigned)ridx << 10) + jj];
    float ld = pos[((unsigned)leaderIdx << 10) + jj];

    float explore = fabsf(xr - Aco * fabsf(Cco * xr - xx));
    float exploit = fabsf(ld - Aco * fabsf(Cco * ld - xx));
    float spiral  = fabsf(fabsf(ld - xx) * __expf(lp) *
                          __cosf(6.28318530718f * lp) + ld);

    float v = (p < 0.5f) ? ((fabsf(Aco) >= 1.0f) ? explore : exploit) : spiral;
    v = fminf(fmaxf(v, lower), upper);
    npos[idx] = v;
}

// ---------------------------------------------------------------------------
// Launch: GEMM1(fused support) -> minmax -> 10x whale -> GEMM2(fused output)
// ---------------------------------------------------------------------------
extern "C" void kernel_launch(void* const* d_in, const int* in_sizes, int n_in,
                              void* d_out, int out_size, void* d_ws, size_t ws_size,
                              hipStream_t stream)
{
    const int Nrow = 8192, Fin = 1024, Fout = 1024, MAXIT = 10;

    const float* input  = (const float*)d_in[0];   // [8192,1024]
    const float* adj    = (const float*)d_in[1];   // [8192,8192]
    const float* h0     = (const float*)d_in[2];   // [8192,1024]
    const float* weight = (const float*)d_in[3];   // [1024,1024]
    const float* lamda  = (const float*)d_in[4];   // scalar
    const float* alpha  = (const float*)d_in[5];   // scalar
    const int*   lval   = (const int*)d_in[6];     // scalar

    // workspace layout
    char* ws = (char*)d_ws;
    const size_t matBytes = (size_t)Nrow * Fin * sizeof(float);   // 32 MB
    float* support = (float*)(ws);                                // [8192,1024]
    float* posA    = (float*)(ws + matBytes);                     // ping
    float* posB    = (float*)(ws + 2 * matBytes);                 // pong
    float* pmin    = (float*)(ws + 3 * matBytes);                 // 512 floats
    float* pmax    = pmin + 512;
    float* bounds  = pmax + 512;                                  // 2 floats

    dim3 blk(256);

    // GEMM1: support = (1-alpha)*(adj@input) + alpha*h0 ; also seeds posA
    {
        dim3 grid(Nrow / TILE_M, Fin / TILE_N);   // 64 x 8
        gemm_bf16_wmma<true><<<grid, blk, 0, stream>>>(
            adj, input, h0, support, posA,
            Nrow, Fin, Nrow, alpha, lval, /*mode=*/0);
    }

    // clip bounds from initial positions
    {
        int n = Nrow * Fin;
        minmax_part<<<512, blk, 0, stream>>>(support, n, pmin, pmax);
        minmax_final<<<1, blk, 0, stream>>>(pmin, pmax, 512, bounds);
    }

    // deterministic host-side leader indices (stateless per call)
    unsigned s = 42u;
    int leaders[MAXIT];
    for (int t = 0; t < MAXIT; ++t) {
        s = s * 1664525u + 1013904223u;
        leaders[t] = (int)(s % (unsigned)Nrow);
    }

    // 10 sequential whale iterations, ping-pong posA <-> posB
    {
        int nblk = (Nrow * Fin) / 256;            // 32768
        float* cur = posA;
        float* nxt = posB;
        for (int t = 0; t < MAXIT; ++t) {
            whale_step<<<nblk, blk, 0, stream>>>(
                cur, nxt, bounds, t, leaders[t], MAXIT);
            float* tmp = cur; cur = nxt; nxt = tmp;
        }
        // after 10 steps result is in posA (cur)
        // GEMM2: out = theta*(opt@weight) + (1-theta)*support
        dim3 grid(Nrow / TILE_M, Fout / TILE_N);  // 64 x 8
        gemm_bf16_wmma<false><<<grid, blk, 0, stream>>>(
            cur, weight, support, (float*)d_out, nullptr,
            Nrow, Fout, Fin, lamda, lval, /*mode=*/1);
    }
    (void)in_sizes; (void)n_in; (void)out_size; (void)ws_size;
}
